// AttnBlock_30580167147897
// MI455X (gfx1250) — matrix operand order repaired and checked by reading, not compiled, not run on hardware
//
#include <hip/hip_runtime.h>
#include <hip/hip_bf16.h>

typedef __attribute__((ext_vector_type(16))) __bf16 v16bf;
typedef __attribute__((ext_vector_type(8)))  __bf16 v8bf;
typedef __attribute__((ext_vector_type(8)))  float  v8f;

#define DEVINL __device__ __forceinline__

// ---- helpers -------------------------------------------------------------

DEVINL __bf16 f2bf(float f) {
  unsigned u = __builtin_bit_cast(unsigned, f);
  unsigned r = (u + 0x7FFFu + ((u >> 16) & 1u)) >> 16;
  return __builtin_bit_cast(__bf16, (unsigned short)r);
}

DEVINL v8f wmma_bf16(v16bf a, v16bf b, v8f c) {
  return __builtin_amdgcn_wmma_f32_16x16x32_bf16(
      /*neg_a=*/false, a, /*neg_b=*/false, b,
      /*c_mod=*/(short)0, c, /*reuse_a=*/false, /*reuse_b=*/false);
}

// A fragment 16x32 bf16 (row-major source, stride in elements).
// Lane: m=lane&15, g=lane>>4. elems 0..15 -> K=16g+0..15.
// Requires (base + m*stride + 16g) 16-byte aligned (true for all uses below).
DEVINL v16bf ldA(const __bf16* base, long stride, int lane) {
  int m = lane & 15, g = lane >> 4;
  const __bf16* p = base + (long)m * stride + 16 * g;
  v8bf lo = *(const v8bf*)p;
  v8bf hi = *(const v8bf*)(p + 8);
  v16bf r;
#pragma unroll
  for (int i = 0; i < 8; ++i) { r[i] = lo[i]; r[i + 8] = hi[i]; }
  return r;
}

// B fragment 32x16 bf16 from a K-contiguous layout: element e -> K=16g+e,
// N = lane&15 selects the row (stride = elements between N rows). 32B aligned.
DEVINL v16bf ldB(const __bf16* base, long stride, int lane) {
  int n = lane & 15, g = lane >> 4;
  return *(const v16bf*)(base + (long)n * stride + 16 * g);
}

// ---- kernel 0: f32 -> bf16 weight conversion -----------------------------

__global__ void w2bf_kernel(const float* __restrict__ w, __bf16* __restrict__ o,
                            int n) {
  for (int i = blockIdx.x * 256 + threadIdx.x; i < n; i += gridDim.x * 256)
    o[i] = f2bf(w[i]);
}

// ---- kernel 1: GroupNorm -> bf16 hT [B,S,C] ------------------------------

__global__ void gn_kernel(const float* __restrict__ x,
                          const float* __restrict__ gamma,
                          const float* __restrict__ beta,
                          __bf16* __restrict__ hT) {
  const int S = 4096, C = 512;
  const int GE = 16 * S;  // elements per (batch, group)
  long base = (long)blockIdx.x * GE;          // x offset of this (b, group)
  int b = blockIdx.x >> 5;
  int ch0 = (blockIdx.x & 31) * 16;           // first channel of group
  __shared__ float s1[256], s2[256];
  float sum = 0.f, sq = 0.f;
  for (int i = threadIdx.x; i < GE; i += 256) {
    float v = x[base + i];
    sum += v; sq += v * v;
  }
  s1[threadIdx.x] = sum; s2[threadIdx.x] = sq;
  __syncthreads();
  for (int off = 128; off > 0; off >>= 1) {
    if ((int)threadIdx.x < off) {
      s1[threadIdx.x] += s1[threadIdx.x + off];
      s2[threadIdx.x] += s2[threadIdx.x + off];
    }
    __syncthreads();
  }
  float mean = s1[0] * (1.f / (float)GE);
  float var  = s2[0] * (1.f / (float)GE) - mean * mean;
  float rstd = rsqrtf(var + 1e-6f);
  float ga[16], be[16];
#pragma unroll
  for (int c = 0; c < 16; ++c) { ga[c] = gamma[ch0 + c] * rstd; be[c] = beta[ch0 + c]; }
  __bf16* hb = hT + (long)b * S * C;
  for (int s = threadIdx.x; s < S; s += 256) {
    v16bf hv;
#pragma unroll
    for (int c = 0; c < 16; ++c) {
      float v = (x[base + (long)c * S + s] - mean) * ga[c] + be[c];
      hv[c] = f2bf(v);
    }
    *(v16bf*)(hb + (long)s * C + ch0) = hv;  // 32B aligned (ch0 mult of 16)
  }
}

// ---- kernel 2/4: LDS-free WMMA GEMM --------------------------------------
// out[o,s] = Wbf[o,:] . hT[s,:] + bias[o]   (contraction over C, K-contig both)
// mode 0: bf16 out [C,S];  mode 1: bf16 out transposed [S,C], *scale;
// mode 2: f32 out [C,S] + residual.

__global__ void __launch_bounds__(256) gemm_kernel(
    const __bf16* __restrict__ Wbf, const __bf16* __restrict__ hT,
    const float* __restrict__ bias, const float* __restrict__ resid,
    float scale, void* __restrict__ outp, int mode) {
  const int C = 512, S = 4096;
  int b = blockIdx.z;
  const __bf16* hb = hT + (long)b * S * C;
  int m0 = blockIdx.y * 128, n0 = blockIdx.x * 128;
  int tid = threadIdx.x, lane = tid & 31, w = tid >> 5;
  int wm = m0 + (w >> 2) * 64;  // wave rows (out channels), 2 waves in M
  int wn = n0 + (w & 3) * 32;   // wave cols (spatial), 4 waves in N
  v8f acc[4][2] = {};

  for (int k0 = 0; k0 < C; k0 += 32) {
    v16bf bf0 = ldB(hb + (long)wn * C + k0, C, lane);
    v16bf bf1 = ldB(hb + (long)(wn + 16) * C + k0, C, lane);
#pragma unroll
    for (int mt = 0; mt < 4; ++mt) {
      v16bf af = ldA(Wbf + (long)(wm + mt * 16) * C + k0, C, lane);
      acc[mt][0] = wmma_bf16(af, bf0, acc[mt][0]);
      acc[mt][1] = wmma_bf16(af, bf1, acc[mt][1]);
    }
  }

  int g = lane >> 4, n = lane & 15;
#pragma unroll
  for (int mt = 0; mt < 4; ++mt)
#pragma unroll
    for (int nt = 0; nt < 2; ++nt)
#pragma unroll
      for (int j = 0; j < 8; ++j) {
        int row = wm + mt * 16 + j + 8 * g;  // output channel
        int col = wn + nt * 16 + n;          // spatial index
        float vl = acc[mt][nt][j] + bias[row];
        if (mode == 0) {
          ((__bf16*)outp)[(long)b * C * S + (long)row * S + col] = f2bf(vl);
        } else if (mode == 1) {
          ((__bf16*)outp)[(long)b * S * C + (long)col * C + row] = f2bf(vl * scale);
        } else {
          long idx = (long)b * C * S + (long)row * S + col;
          ((float*)outp)[idx] = vl + resid[idx];
        }
      }
}

// ---- kernel 3: flash attention ------------------------------------------
// qT,kT: [B,S,C] bf16 (q pre-scaled); v: [B,C,S] bf16; out: [B,S,C] bf16.

__global__ void __launch_bounds__(256) attn_kernel(
    const __bf16* __restrict__ qT, const __bf16* __restrict__ kT,
    const __bf16* __restrict__ vm, __bf16* __restrict__ omT) {
  const int C = 512, S = 4096, QB = 64, KB = 128;
  int b = blockIdx.x >> 6;
  int i0 = (blockIdx.x & 63) * QB;
  const __bf16* qb = qT + (long)b * S * C;
  const __bf16* kb = kT + (long)b * S * C;
  const __bf16* vb = vm + (long)b * C * S;
  __bf16* ob = omT + (long)b * S * C;

  __shared__ float  s_sc[QB][KB];  // raw scores (f32)      32 KB
  __shared__ __bf16 s_p[QB][KB];   // exp(scores) (bf16)    16 KB
  __shared__ float  s_m[QB], s_l[QB], s_alpha[QB], s_mn[QB];
  __shared__ float  s_red[QB][4];

  int tid = threadIdx.x, lane = tid & 31, w = tid >> 5;
  int g = lane >> 4, n = lane & 15;
  int sr = (w >> 1) * 16;   // score phase: row strip (queries)
  int sc = (w & 1) * 64;    // score phase: key-column half
  int c0 = w * 64;          // output phase: channel slice

  if (tid < QB) { s_m[tid] = -3.0e38f; s_l[tid] = 0.f; }
  __syncthreads();

  v8f oacc[4][4] = {};  // [query 16-row tile][channel 16-col tile]

  for (int j0 = 0; j0 < S; j0 += KB) {
    // prefetch next key tile into cache while we compute on this one
    if (j0 + KB < S) {
      const __bf16* nk = kb + (long)(j0 + KB) * C;
#pragma unroll
      for (int t = 0; t < 4; ++t)
        __builtin_prefetch(nk + ((long)tid * 4 + t) * 64, 0, 1);
    }
    // ---- scores: S1[64,128] = Q^T K (contraction over C) ----
    v8f sacc[4] = {};
    for (int k0 = 0; k0 < C; k0 += 32) {
      v16bf af = ldA(qb + (long)(i0 + sr) * C + k0, C, lane);
#pragma unroll
      for (int nt = 0; nt < 4; ++nt) {
        v16bf bf = ldB(kb + (long)(j0 + sc + nt * 16) * C + k0, C, lane);
        sacc[nt] = wmma_bf16(af, bf, sacc[nt]);
      }
    }
#pragma unroll
    for (int nt = 0; nt < 4; ++nt)
#pragma unroll
      for (int j = 0; j < 8; ++j)
        s_sc[sr + j + 8 * g][sc + nt * 16 + n] = sacc[nt][j];
    __syncthreads();

    // ---- online softmax: 4 threads per row, 32 cols each ----
    {
      int row = tid >> 2, q4 = (tid & 3) * 32;
      float mx = -3.0e38f;
#pragma unroll
      for (int jj = 0; jj < 32; ++jj) mx = fmaxf(mx, s_sc[row][q4 + jj]);
      s_red[row][tid & 3] = mx;
    }
    __syncthreads();
    if (tid < QB) {
      float mj = fmaxf(fmaxf(s_red[tid][0], s_red[tid][1]),
                       fmaxf(s_red[tid][2], s_red[tid][3]));
      float mo = s_m[tid];
      float mn = fmaxf(mo, mj);
      s_mn[tid] = mn;
      s_alpha[tid] = __expf(mo - mn);
      s_m[tid] = mn;
    }
    __syncthreads();
    {
      int row = tid >> 2, q4 = (tid & 3) * 32;
      float mn = s_mn[row];
      float sm = 0.f;
#pragma unroll
      for (int jj = 0; jj < 32; ++jj) {
        float e = __expf(s_sc[row][q4 + jj] - mn);
        s_p[row][q4 + jj] = f2bf(e);
        sm += e;
      }
      s_red[row][tid & 3] = sm;
    }
    __syncthreads();
    if (tid < QB) {
      s_l[tid] = s_l[tid] * s_alpha[tid] +
                 (s_red[tid][0] + s_red[tid][1] + s_red[tid][2] + s_red[tid][3]);
    }
    __syncthreads();

    // ---- rescale running output by alpha(row) ----
#pragma unroll
    for (int mt = 0; mt < 4; ++mt)
#pragma unroll
      for (int j = 0; j < 8; ++j) {
        float a = s_alpha[mt * 16 + j + 8 * g];
#pragma unroll
        for (int nt = 0; nt < 4; ++nt) oacc[mt][nt][j] *= a;
      }

    // ---- O[64,64ch] += P[64,128] x V'[128,64ch] ----
    for (int kk = 0; kk < KB; kk += 32) {
      v16bf bfr[4];
#pragma unroll
      for (int nt = 0; nt < 4; ++nt)
        bfr[nt] = ldB(vb + (long)(c0 + nt * 16) * S + (j0 + kk), S, lane);
#pragma unroll
      for (int mt = 0; mt < 4; ++mt) {
        v16bf pa = ldA(&s_p[mt * 16][0] + kk, KB, lane);
#pragma unroll
        for (int nt = 0; nt < 4; ++nt)
          oacc[mt][nt] = wmma_bf16(pa, bfr[nt], oacc[mt][nt]);
      }
    }
    __syncthreads();  // protect s_sc / s_p for next key tile
  }

  if (tid < QB) s_alpha[tid] = 1.0f / s_l[tid];
  __syncthreads();
  // store transposed [S,C]: lanes n contiguous over channels -> 32B runs
#pragma unroll
  for (int mt = 0; mt < 4; ++mt)
#pragma unroll
    for (int j = 0; j < 8; ++j) {
      float inv = s_alpha[mt * 16 + j + 8 * g];
#pragma unroll
      for (int nt = 0; nt < 4; ++nt)
        ob[(long)(i0 + mt * 16 + j + 8 * g) * C + (c0 + nt * 16 + n)] =
            f2bf(oacc[mt][nt][j] * inv);
    }
}

// ---- host launcher -------------------------------------------------------

extern "C" void kernel_launch(void* const* d_in, const int* in_sizes, int n_in,
                              void* d_out, int out_size, void* d_ws, size_t ws_size,
                              hipStream_t stream) {
  (void)in_sizes; (void)n_in; (void)out_size; (void)ws_size;
  const float* x     = (const float*)d_in[0];
  const float* gamma = (const float*)d_in[1];
  const float* beta  = (const float*)d_in[2];
  const float* wq    = (const float*)d_in[3];
  const float* bq    = (const float*)d_in[4];
  const float* wk    = (const float*)d_in[5];
  const float* bk    = (const float*)d_in[6];
  const float* wv    = (const float*)d_in[7];
  const float* bv    = (const float*)d_in[8];
  const float* wo    = (const float*)d_in[9];
  const float* bo    = (const float*)d_in[10];
  float* out = (float*)d_out;

  const long BCS = 4L * 512 * 4096;
  const int  CC  = 512 * 512;
  __bf16* hT  = (__bf16*)d_ws;   // 16 MB  [B,S,C] normalized input
  __bf16* qT  = hT + BCS;        // 16 MB  [B,S,C], pre-scaled by C^-0.5
  __bf16* kT  = qT + BCS;        // 16 MB  [B,S,C]
  __bf16* vB  = kT + BCS;        // 16 MB  [B,C,S]
  __bf16* wqb = vB + BCS;        // 512 KB each, bf16 weights
  __bf16* wkb = wqb + CC;
  __bf16* wvb = wkb + CC;
  __bf16* wob = wvb + CC;
  __bf16* hatT = hT;             // reuse hT for attention output [B,S,C]

  w2bf_kernel<<<256, 256, 0, stream>>>(wq, wqb, CC);
  w2bf_kernel<<<256, 256, 0, stream>>>(wk, wkb, CC);
  w2bf_kernel<<<256, 256, 0, stream>>>(wv, wvb, CC);
  w2bf_kernel<<<256, 256, 0, stream>>>(wo, wob, CC);

  gn_kernel<<<128, 256, 0, stream>>>(x, gamma, beta, hT);

  const float scale = 0.04419417382415922f;  // 512^-0.5
  dim3 gg(32, 4, 4);  // (N tiles, M tiles, batch)
  gemm_kernel<<<gg, 256, 0, stream>>>(wqb, hT, bq, nullptr, scale, qT, 1);
  gemm_kernel<<<gg, 256, 0, stream>>>(wkb, hT, bk, nullptr, 1.0f,  kT, 1);
  gemm_kernel<<<gg, 256, 0, stream>>>(wvb, hT, bv, nullptr, 1.0f,  vB, 0);

  attn_kernel<<<256, 256, 0, stream>>>(qT, kT, vB, hatT);

  gemm_kernel<<<gg, 256, 0, stream>>>(wob, hatT, bo, x, 1.0f, out, 2);
}
